// Net_26620207301223
// MI455X (gfx1250) — matrix operand order repaired and checked by reading, not compiled, not run on hardware
//
#include <hip/hip_runtime.h>

#define N_NODES 8192
#define DIM 512
#define N_EDGES 262144

typedef __attribute__((ext_vector_type(16))) __bf16 v16bf;
typedef __attribute__((ext_vector_type(8)))  float  v8f;

union BFrag {
    v16bf v;
    uint4 q[2];
    unsigned short us[16];
};

__device__ __forceinline__ unsigned short f32_to_bf16_bits(float f) {
    unsigned u = __float_as_uint(f);
    unsigned r = u + 0x7FFFu + ((u >> 16) & 1u);   // round-to-nearest-even
    return (unsigned short)(r >> 16);
}
__device__ __forceinline__ float bf16_bits_to_f32(unsigned short s) {
    return __uint_as_float(((unsigned)s) << 16);
}

// ---------------------------------------------------------------- utility
__global__ void zero_f32_kernel(float* __restrict__ p, size_t n) {
    size_t i = (size_t)blockIdx.x * blockDim.x + threadIdx.x;
    if (i < n) p[i] = 0.0f;
}

__global__ void cvt_f32_to_bf16_kernel(const float* __restrict__ in,
                                       unsigned short* __restrict__ out, size_t n) {
    size_t i = (size_t)blockIdx.x * blockDim.x + threadIdx.x;
    if (i < n) out[i] = f32_to_bf16_bits(in[i]);
}

// ---------------------------------------------------------------- aggregation
__global__ void deg_kernel(const int* __restrict__ dst, float* __restrict__ deg, int E) {
    int e = blockIdx.x * blockDim.x + threadIdx.x;
    if (e < E) atomicAdd(&deg[dst[e]], 1.0f);
}

__global__ __launch_bounds__(256) void scatter_kernel(const int* __restrict__ src,
                                                      const int* __restrict__ dst,
                                                      const float* __restrict__ x,
                                                      float* __restrict__ msg) {
    int e    = blockIdx.x * 8 + (threadIdx.x >> 5);
    int lane = threadIdx.x & 31;
    int s = src[e], d = dst[e];
    const float* xs = x   + (size_t)s * DIM;
    float*       ms = msg + (size_t)d * DIM;
#pragma unroll
    for (int f = lane; f < DIM; f += 32) atomicAdd(ms + f, xs[f]);
}

// msg / max(deg,1), converted to bf16
__global__ void agg_finalize_kernel(const float* __restrict__ msg,
                                    const float* __restrict__ deg,
                                    unsigned short* __restrict__ out) {
    size_t i = (size_t)blockIdx.x * blockDim.x + threadIdx.x;
    if (i < (size_t)N_NODES * DIM) {
        float dv = fmaxf(deg[i >> 9], 1.0f);   // D = 512 = 2^9
        out[i] = f32_to_bf16_bits(msg[i] / dv);
    }
}

// ---------------------------------------------------------------- weight pre-pack
// Pack fp32 W[K,Nc] (row-major) into WMMA B-fragment order:
// tile t=(kt*nTiles+nt): 32 lanes x 16 contiguous bf16 per lane.
// lane<16 : col=nt*16+lane,     K = kt*32 + 0..7, 16..23
// lane>=16: col=nt*16+lane-16,  K = kt*32 + 8..15, 24..31
__global__ void pack_b_kernel(const float* __restrict__ W, unsigned short* __restrict__ out,
                              int K, int Nc) {
    int t = blockIdx.x * blockDim.x + threadIdx.x;
    int nTiles = Nc >> 4;
    int total  = (K >> 5) * nTiles * 32;
    if (t >= total) return;
    int lane = t & 31;
    int tile = t >> 5;
    int kt = tile / nTiles, nt = tile % nTiles;
    int col = nt * 16 + (lane & 15);
    int k0  = kt * 32 + ((lane >> 4) << 3);
    unsigned short* o = out + (size_t)t * 16;
#pragma unroll
    for (int j = 0; j < 16; ++j)
        o[j] = f32_to_bf16_bits(W[(size_t)(k0 + j + ((j >> 3) << 3)) * Nc + col]);
}

// ---------------------------------------------------------------- WMMA GEMM
// Out[M,Nc] = relu( A@B (+ A2@B2) + bias ),  bf16 in / f32 acc / bf16 out.
// One wave -> 16(M) x 16*NSUB(N) output tile: A fragment loaded once,
// reused by NSUB back-to-back WMMAs. 8 waves per 256-thread block.
template <int NSUB>
__global__ __launch_bounds__(256) void gemm_bf16_wmma(
    const unsigned short* __restrict__ A,   const unsigned short* __restrict__ Bpk,
    const unsigned short* __restrict__ A2,  const unsigned short* __restrict__ B2pk,
    const float* __restrict__ bias, unsigned short* __restrict__ Out,
    int M, int K, int Nc, int doRelu)
{
    const int lane    = threadIdx.x & 31;
    const int wave    = threadIdx.x >> 5;
    const int nTiles  = Nc >> 4;            // 16-wide N tiles
    const int nGroups = nTiles / NSUB;      // per-wave N groups
    const int waveId  = blockIdx.x * 8 + wave;
    const int tileM   = waveId / nGroups;
    const int ntBase  = (waveId % nGroups) * NSUB;
    if (tileM * 16 >= M) return;            // wave-uniform; EXEC stays all-1s

    const int row  = tileM * 16 + (lane & 15);
    const int koff = (lane >> 4) << 3;      // lanes 16-31 shift K by 8
    const int kTiles = K >> 5;
    const size_t bStride = (size_t)nTiles * 512;   // 32 lanes * 16 elems per k-tile row

    v8f acc[NSUB];
#pragma unroll
    for (int s = 0; s < NSUB; ++s) acc[s] = (v8f){};

    {
        const unsigned short* aRow = A + (size_t)row * K + koff;
        const unsigned short* bPtr = Bpk + ((size_t)ntBase * 32 + lane) * 16;
        for (int kt = 0; kt < kTiles; ++kt) {
            BFrag a;
            const uint4* ap = (const uint4*)(aRow + kt * 32);
            a.q[0] = ap[0];            // K +0..7   (16 bytes)
            a.q[1] = ap[2];            // K +16..23 (skip 8 elems)
#pragma unroll
            for (int s = 0; s < NSUB; ++s) {
                BFrag b;
                const uint4* bp = (const uint4*)(bPtr + (size_t)s * 512);
                b.q[0] = bp[0];
                b.q[1] = bp[1];
                acc[s] = __builtin_amdgcn_wmma_f32_16x16x32_bf16(
                    false, a.v, false, b.v, (short)0, acc[s], false, false);
            }
            bPtr += bStride;
        }
    }
    if (A2 != nullptr) {
        const unsigned short* aRow = A2 + (size_t)row * K + koff;
        const unsigned short* bPtr = B2pk + ((size_t)ntBase * 32 + lane) * 16;
        for (int kt = 0; kt < kTiles; ++kt) {
            BFrag a;
            const uint4* ap = (const uint4*)(aRow + kt * 32);
            a.q[0] = ap[0];
            a.q[1] = ap[2];
#pragma unroll
            for (int s = 0; s < NSUB; ++s) {
                BFrag b;
                const uint4* bp = (const uint4*)(bPtr + (size_t)s * 512);
                b.q[0] = bp[0];
                b.q[1] = bp[1];
                acc[s] = __builtin_amdgcn_wmma_f32_16x16x32_bf16(
                    false, a.v, false, b.v, (short)0, acc[s], false, false);
            }
            bPtr += bStride;
        }
    }

    // C/D layout: VGPR r -> M = tileM*16 + r (+8 for lanes 16-31), N = 16*nt + (lane&15)
    const int rbase = tileM * 16 + ((lane >> 4) << 3);
#pragma unroll
    for (int s = 0; s < NSUB; ++s) {
        const int   col = (ntBase + s) * 16 + (lane & 15);
        const float bv  = bias ? bias[col] : 0.0f;
#pragma unroll
        for (int r = 0; r < 8; ++r) {
            float v = acc[s][r] + bv;
            if (doRelu) v = fmaxf(v, 0.0f);
            Out[(size_t)(rbase + r) * Nc + col] = f32_to_bf16_bits(v);
        }
    }
}

// ---------------------------------------------------------------- final 64->3 layer
__global__ void final_layer_kernel(const unsigned short* __restrict__ h4,
                                   const float* __restrict__ W3, const float* __restrict__ b3,
                                   float* __restrict__ y, float* __restrict__ sq) {
    int i = blockIdx.x * blockDim.x + threadIdx.x;
    if (i >= N_NODES) return;
    float o0 = b3[0], o1 = b3[1], o2 = b3[2];
    const unsigned short* r = h4 + (size_t)i * 64;
#pragma unroll
    for (int k = 0; k < 64; ++k) {
        float v = bf16_bits_to_f32(r[k]);
        o0 = fmaf(v, W3[k * 3 + 0], o0);
        o1 = fmaf(v, W3[k * 3 + 1], o1);
        o2 = fmaf(v, W3[k * 3 + 2], o2);
    }
    y[i * 3 + 0] = o0; y[i * 3 + 1] = o1; y[i * 3 + 2] = o2;
    sq[i] = o0 * o0 + o1 * o1 + o2 * o2;
}

// ---------------------------------------------------------------- cdist (256 MB out, NT stores)
__global__ __launch_bounds__(256) void cdist_kernel(const float* __restrict__ y,
                                                    const float* __restrict__ sq,
                                                    float* __restrict__ out) {
    int j = blockIdx.x * 256 + threadIdx.x;
    int i = blockIdx.y;
    float dot = y[3 * i + 0] * y[3 * j + 0]
              + y[3 * i + 1] * y[3 * j + 1]
              + y[3 * i + 2] * y[3 * j + 2];
    float d2 = sq[i] + sq[j] - 2.0f * dot;
    float d  = (d2 > 0.0f) ? sqrtf(d2) : 0.0f;
    __builtin_nontemporal_store(d, &out[(size_t)i * N_NODES + j]);
}

// ---------------------------------------------------------------- launch
extern "C" void kernel_launch(void* const* d_in, const int* in_sizes, int n_in,
                              void* d_out, int out_size, void* d_ws, size_t ws_size,
                              hipStream_t stream) {
    const float* x   = (const float*)d_in[0];
    const int*   ei  = (const int*)d_in[1];           // [2,E] flat: src then dst
    const float* W_l = (const float*)d_in[2];
    const float* b_l = (const float*)d_in[3];
    const float* W_r = (const float*)d_in[4];
    const float* Wa  = (const float*)d_in[5];
    const float* ba  = (const float*)d_in[6];
    const float* W1  = (const float*)d_in[7];
    const float* b1  = (const float*)d_in[8];
    const float* W2  = (const float*)d_in[9];
    const float* b2  = (const float*)d_in[10];
    const float* W3  = (const float*)d_in[11];
    const float* b3  = (const float*)d_in[12];
    const int* src = ei;
    const int* dst = ei + N_EDGES;

    // workspace carve
    char* w = (char*)d_ws;
    auto carve = [&](size_t bytes) -> void* {
        void* p = (void*)w;
        w += (bytes + 255) & ~(size_t)255;
        return p;
    };
    float*          msg   = (float*)carve((size_t)N_NODES * DIM * 4);
    float*          deg   = (float*)carve((size_t)N_NODES * 4);
    unsigned short* xbf   = (unsigned short*)carve((size_t)N_NODES * DIM * 2);
    unsigned short* aggbf = (unsigned short*)carve((size_t)N_NODES * DIM * 2);
    unsigned short* h1    = (unsigned short*)carve((size_t)N_NODES * 512 * 2);
    unsigned short* h2    = (unsigned short*)carve((size_t)N_NODES * 256 * 2);
    unsigned short* h3    = (unsigned short*)carve((size_t)N_NODES * 128 * 2);
    unsigned short* h4    = (unsigned short*)carve((size_t)N_NODES * 64 * 2);
    float*          y     = (float*)carve((size_t)N_NODES * 3 * 4);
    float*          sq    = (float*)carve((size_t)N_NODES * 4);
    unsigned short* WlP   = (unsigned short*)carve((size_t)512 * 512 * 2);
    unsigned short* WrP   = (unsigned short*)carve((size_t)512 * 512 * 2);
    unsigned short* WaP   = (unsigned short*)carve((size_t)512 * 256 * 2);
    unsigned short* W1P   = (unsigned short*)carve((size_t)256 * 128 * 2);
    unsigned short* W2P   = (unsigned short*)carve((size_t)128 * 64 * 2);

    const size_t nd = (size_t)N_NODES * DIM;   // 4,194,304

    // 1) zero accumulators
    zero_f32_kernel<<<(unsigned)((nd + 255) / 256), 256, 0, stream>>>(msg, nd);
    zero_f32_kernel<<<(N_NODES + 255) / 256, 256, 0, stream>>>(deg, (size_t)N_NODES);

    // 2) degree + scatter-add aggregation (L2-resident atomics)
    deg_kernel<<<N_EDGES / 256, 256, 0, stream>>>(dst, deg, N_EDGES);
    scatter_kernel<<<N_EDGES / 8, 256, 0, stream>>>(src, dst, x, msg);

    // 3) bf16 conversions
    cvt_f32_to_bf16_kernel<<<(unsigned)((nd + 255) / 256), 256, 0, stream>>>(x, xbf, nd);
    agg_finalize_kernel<<<(unsigned)((nd + 255) / 256), 256, 0, stream>>>(msg, deg, aggbf);

    // 4) pack weights into WMMA B-fragment order
    pack_b_kernel<<<64, 256, 0, stream>>>(W_l, WlP, 512, 512);   // 16*32*32 threads
    pack_b_kernel<<<64, 256, 0, stream>>>(W_r, WrP, 512, 512);
    pack_b_kernel<<<32, 256, 0, stream>>>(Wa,  WaP, 512, 256);
    pack_b_kernel<<< 8, 256, 0, stream>>>(W1,  W1P, 256, 128);
    pack_b_kernel<<< 2, 256, 0, stream>>>(W2,  W2P, 128,  64);

    // 5) GEMM chain: blocks = (M/16)*(Nc/16/NSUB)/8
    // L1: 512 M-tiles * 8 groups  / 8 = 512 blocks (NSUB=4, fused agg@Wl + x@Wr)
    gemm_bf16_wmma<4><<<512, 256, 0, stream>>>(aggbf, WlP, xbf, WrP, b_l, h1,
                                               N_NODES, 512, 512, 1);
    // L2: 512 * 4 / 8 = 256 blocks (NSUB=4)
    gemm_bf16_wmma<4><<<256, 256, 0, stream>>>(h1, WaP, nullptr, nullptr, ba, h2,
                                               N_NODES, 512, 256, 1);
    // L3: 512 * 4 / 8 = 256 blocks (NSUB=2)
    gemm_bf16_wmma<2><<<256, 256, 0, stream>>>(h2, W1P, nullptr, nullptr, b1, h3,
                                               N_NODES, 256, 128, 1);
    // L4: 512 * 4 / 8 = 256 blocks (NSUB=1, keep occupancy for tiny layer)
    gemm_bf16_wmma<1><<<256, 256, 0, stream>>>(h3, W2P, nullptr, nullptr, b2, h4,
                                               N_NODES, 128, 64, 1);

    // 6) tiny 64->3 layer + squared norms
    final_layer_kernel<<<N_NODES / 256, 256, 0, stream>>>(h4, W3, b3, y, sq);

    // 7) cdist -> d_out (8192x8192 fp32, nontemporal stores)
    dim3 cg(N_NODES / 256, N_NODES);
    cdist_kernel<<<cg, 256, 0, stream>>>(y, sq, (float*)d_out);

    (void)in_sizes; (void)n_in; (void)out_size; (void)ws_size;
}
